// HistoricalPromptDecoder_25348896981519
// MI455X (gfx1250) — matrix operand order repaired and checked by reading, not compiled, not run on hardware
//
#include <hip/hip_runtime.h>
#include <hip/hip_bf16.h>

// ---------------------------------------------------------------------------
// Problem constants (from reference setup_inputs)
// ---------------------------------------------------------------------------
#define BB      4
#define CK      64
#define CV      512
#define NE      23040
#define NQ      576
#define TOPK    20
#define QT      (NQ / 16)          // 36 query tiles of 16
#define NT      (NE / 16)          // 1440 memory-column tiles of 16
#define NSLICES 16                 // NE split into 16 slices for parallel top-k
#define TPS     (NT / NSLICES)     // 90 tiles per slice (even)
#define WAVES   8                  // waves per workgroup in the WMMA kernel

typedef __attribute__((ext_vector_type(16))) __bf16 v16bf;
typedef __attribute__((ext_vector_type(8)))  float  v8f;
typedef __attribute__((ext_vector_type(4)))  float  v4f;

// round-to-nearest-even f32 -> bf16 (raw u16)
__device__ __forceinline__ unsigned short f2bf(float f) {
    unsigned int u = __float_as_uint(f);
    unsigned int r = (u + 0x7FFFu + ((u >> 16) & 1u)) >> 16;
    return (unsigned short)r;
}

// ---------------------------------------------------------------------------
// Register-resident sorted top-20 (descending). Branchless bubble insert:
// each stage is 1 v_cmp + 4 v_cndmask; single threshold guard up front.
// ---------------------------------------------------------------------------
struct TopK {
    float v[TOPK];
    int   i[TOPK];
    __device__ __forceinline__ void init() {
#pragma unroll
        for (int j = 0; j < TOPK; ++j) { v[j] = -3.0e38f; i[j] = 0; }
    }
    __device__ __forceinline__ void insert(float nv, int ni) {
        if (nv <= v[TOPK - 1]) return;     // common fast path: 1 compare
        float cv = nv; int ci = ni;
#pragma unroll
        for (int j = 0; j < TOPK; ++j) {
            bool  sw = cv > v[j];
            float ov = v[j]; int oi = i[j];
            v[j] = sw ? cv : ov;  i[j] = sw ? ci : oi;
            cv   = sw ? ov : cv;  ci   = sw ? oi : ci;
        }
    }
};

// ---------------------------------------------------------------------------
// Kernel 1: pack mk [B,CK,NE] f32 -> bf16 in WMMA A-matrix (16x32) layout.
// A rows = 16 memory columns of a tile. lane L: m = L&15, h = L>>4;
// j = 2g+p -> k = (g<4) ? 8h+2g+p : 16+8h+2(g-4)+p  (+ 32*kchunk).
// ---------------------------------------------------------------------------
__global__ void pack_mk_kernel(const float* __restrict__ mk,
                               unsigned short* __restrict__ aPack) {
    int b  = blockIdx.x / NT;
    int nt = blockIdx.x % NT;
    int t    = threadIdx.x;        // 64 threads: 2 k-chunks x 32 lanes
    int kc   = t >> 5;
    int lane = t & 31;
    int h = lane >> 4, m = lane & 15;
    int n = nt * 16 + m;
    unsigned short* dst = aPack + (((size_t)(b * NT + nt) * 2 + kc) * 32 + lane) * 16;
#pragma unroll
    for (int j = 0; j < 16; ++j) {
        int g = j >> 1, p = j & 1;
        int k = ((g < 4) ? (8 * h + 2 * g + p) : (16 + 8 * h + 2 * (g - 4) + p)) + kc * 32;
        dst[j] = f2bf(mk[((size_t)b * CK + k) * NE + n]);
    }
}

// ---------------------------------------------------------------------------
// Kernel 2: pack qk [B,CK,NQ] f32 -> bf16 in WMMA B-matrix (32x16) layout.
// B columns = 16 queries. lane L: q = L&15, h = L>>4; j -> k = (j<8) ? 8h+j : 16+8h+(j-8).
// ---------------------------------------------------------------------------
__global__ void pack_qk_kernel(const float* __restrict__ qk,
                               unsigned short* __restrict__ bPack) {
    int b  = blockIdx.x / QT;
    int qt = blockIdx.x % QT;
    int t    = threadIdx.x;        // 64 threads: 2 k-chunks x 32 lanes
    int kc   = t >> 5;
    int lane = t & 31;
    int h = lane >> 4, ql = lane & 15;
    int q = qt * 16 + ql;
    unsigned short* dst = bPack + (((size_t)(b * QT + qt) * 2 + kc) * 32 + lane) * 16;
#pragma unroll
    for (int j = 0; j < 16; ++j) {
        int k = kc * 32 + ((j < 8) ? (8 * h + j) : (16 + 8 * h + (j - 8)));
        dst[j] = f2bf(qk[((size_t)b * CK + k) * NQ + q]);
    }
}

// ---------------------------------------------------------------------------
// Kernel 3: column bias = -|mk[:,n]|^2 / sqrt(CK);  affinity = 0.25*ab + bias
// ---------------------------------------------------------------------------
__global__ void bias_kernel(const float* __restrict__ mk, float* __restrict__ bias) {
    int i = blockIdx.x * blockDim.x + threadIdx.x;
    if (i >= BB * NE) return;
    int b = i / NE, n = i % NE;
    float s = 0.f;
#pragma unroll 8
    for (int k = 0; k < CK; ++k) {
        float v = mk[((size_t)b * CK + k) * NE + n];
        s += v * v;
    }
    bias[i] = -s * 0.125f;   // -a_sq / sqrt(64)
}

// ---------------------------------------------------------------------------
// Kernel 4: fused WMMA affinity + per-slice top-20, fully in registers.
// D = mk_tile(16m x 64k) x qk_tile(64k x 16q): lane = query, acc VGPR r holds
// memory row (r + 8*(lane>>4)). Each lane keeps an exact top-20 of the memory
// rows it sees; 2 lanes x 8 waves = 16 disjoint sub-streams per query, merged
// once at the end through LDS. No LDS traffic in the hot loop.
// grid = B * QT * NSLICES, block = 256 (8 waves). Two tiles per iteration.
// ---------------------------------------------------------------------------
__global__ __launch_bounds__(256) void aff_topk_kernel(
    const unsigned short* __restrict__ aPackRaw,
    const unsigned short* __restrict__ bPackRaw,
    const float* __restrict__ bias,
    float* __restrict__ candVal, int* __restrict__ candIdx) {

    __shared__ float mVal[WAVES * 32][TOPK];   // 20 KB
    __shared__ int   mIdx[WAVES * 32][TOPK];   // 20 KB

    int s  = blockIdx.x % NSLICES;
    int qt = (blockIdx.x / NSLICES) % QT;
    int b  = blockIdx.x / (NSLICES * QT);
    int lane = threadIdx.x & 31;
    int wave = threadIdx.x >> 5;
    int h    = lane >> 4;

    // B tile (qk, K=64 x 16 queries): loaded once, reused for the whole slice
    const v16bf* bp = reinterpret_cast<const v16bf*>(
        bPackRaw + ((size_t)(b * QT + qt) * 2) * 32 * 16);
    v16bf qb0 = bp[lane];        // K = 0..31
    v16bf qb1 = bp[32 + lane];   // K = 32..63

    const float* biasb = bias + (size_t)b * NE;

    TopK tk; tk.init();

    for (int t0 = wave * 2; t0 < TPS; t0 += WAVES * 2) {
        int nt0 = s * TPS + t0;                       // this wave's tile pair
        const v16bf* ap = reinterpret_cast<const v16bf*>(
            aPackRaw + ((size_t)(b * NT + nt0) * 2) * 32 * 16);
        // tile nt0: chunks at [0,64); tile nt0+1 contiguous at [64,128)
        v16bf a00 = ap[lane];
        v16bf a01 = ap[32 + lane];
        v16bf a10 = ap[64 + lane];
        v16bf a11 = ap[96 + lane];

        int m0 = nt0 * 16 + 8 * h;        // first memory row held by this lane
        int m1 = m0 + 16;                 // same for the second tile

        // bias as explicit 128-bit vector loads (m0 is 8-float aligned)
        const v4f* bq0 = reinterpret_cast<const v4f*>(biasb + m0);
        const v4f* bq1 = reinterpret_cast<const v4f*>(biasb + m1);
        v4f bv00 = bq0[0], bv01 = bq0[1];
        v4f bv10 = bq1[0], bv11 = bq1[1];

        if (t0 + 2 * WAVES < TPS)   // prefetch next pair -> global_prefetch_b8
            __builtin_prefetch((const void*)(ap + 128 * WAVES), 0, 1);

        v8f z = {0.f, 0.f, 0.f, 0.f, 0.f, 0.f, 0.f, 0.f};
        v8f acc0 = __builtin_amdgcn_wmma_f32_16x16x32_bf16(false, a00, false, qb0,
                                                           (short)0, z,    false, false);
        acc0     = __builtin_amdgcn_wmma_f32_16x16x32_bf16(false, a01, false, qb1,
                                                           (short)0, acc0, false, false);
        v8f acc1 = __builtin_amdgcn_wmma_f32_16x16x32_bf16(false, a10, false, qb0,
                                                           (short)0, z,    false, false);
        acc1     = __builtin_amdgcn_wmma_f32_16x16x32_bf16(false, a11, false, qb1,
                                                           (short)0, acc1, false, false);

#pragma unroll
        for (int r = 0; r < 4; ++r)
            tk.insert(0.25f * acc0[r] + bv00[r], m0 + r);
#pragma unroll
        for (int r = 0; r < 4; ++r)
            tk.insert(0.25f * acc0[4 + r] + bv01[r], m0 + 4 + r);
#pragma unroll
        for (int r = 0; r < 4; ++r)
            tk.insert(0.25f * acc1[r] + bv10[r], m1 + r);
#pragma unroll
        for (int r = 0; r < 4; ++r)
            tk.insert(0.25f * acc1[4 + r] + bv11[r], m1 + 4 + r);
    }

    // dump all 256 lane-local lists; lanes with (lane&15)==q hold query q data
#pragma unroll
    for (int j = 0; j < TOPK; ++j) {
        mVal[wave * 32 + lane][j] = tk.v[j];
        mIdx[wave * 32 + lane][j] = tk.i[j];
    }
    __syncthreads();

    // merge the 16 sub-stream lists per query -> workgroup top-20
    if (threadIdx.x < 16) {
        int q = threadIdx.x;
        TopK f; f.init();
        for (int w = 0; w < WAVES; ++w) {
#pragma unroll 2
            for (int half = 0; half < 2; ++half) {
                int src = w * 32 + q + 16 * half;
                for (int c = 0; c < TOPK; ++c)
                    f.insert(mVal[src][c], mIdx[src][c]);
            }
        }
        size_t base = (((size_t)b * NQ + qt * 16 + q) * NSLICES + s) * TOPK;
#pragma unroll
        for (int j = 0; j < TOPK; ++j) { candVal[base + j] = f.v[j]; candIdx[base + j] = f.i[j]; }
    }
}

// ---------------------------------------------------------------------------
// Kernel 5: merge per-slice candidates -> global top-20, softmax, gather mv.
// grid = B*NQ, block = 256. out layout [B, CV, NQ].
// ---------------------------------------------------------------------------
__global__ __launch_bounds__(256) void readout_kernel(
    const float* __restrict__ candVal, const int* __restrict__ candIdx,
    const float* __restrict__ mv, float* __restrict__ out) {

    __shared__ float wsm[TOPK];
    __shared__ int   widx[TOPK];

    int q = blockIdx.x % NQ;
    int b = blockIdx.x / NQ;

    if (threadIdx.x == 0) {
        TopK f; f.init();
        size_t base = ((size_t)b * NQ + q) * NSLICES * TOPK;
        for (int c = 0; c < NSLICES * TOPK; ++c)
            f.insert(candVal[base + c], candIdx[base + c]);
        float m = f.v[0];   // sorted descending -> max
        float e[TOPK]; float sum = 0.f;
#pragma unroll
        for (int j = 0; j < TOPK; ++j) { e[j] = __expf(f.v[j] - m); sum += e[j]; }
        float inv = 1.f / sum;
#pragma unroll
        for (int j = 0; j < TOPK; ++j) { wsm[j] = e[j] * inv; widx[j] = f.i[j]; }
    }
    __syncthreads();

    for (int c = threadIdx.x; c < CV; c += blockDim.x) {
        const float* mvb = mv + ((size_t)b * CV + c) * NE;
        float acc = 0.f;
#pragma unroll
        for (int j = 0; j < TOPK; ++j) acc += wsm[j] * mvb[widx[j]];
        out[((size_t)b * CV + c) * NQ + q] = acc;
    }
}

// ---------------------------------------------------------------------------
// Launch
// ---------------------------------------------------------------------------
extern "C" void kernel_launch(void* const* d_in, const int* in_sizes, int n_in,
                              void* d_out, int out_size, void* d_ws, size_t ws_size,
                              hipStream_t stream) {
    const float* mk = (const float*)d_in[0];
    const float* qk = (const float*)d_in[1];
    const float* mv = (const float*)d_in[2];
    float* out = (float*)d_out;

    char* ws = (char*)d_ws;
    size_t off = 0;
    unsigned short* aPack = (unsigned short*)(ws + off);       // mk, bf16 A-layout
    off += (size_t)BB * NT * 2 * 32 * 16 * sizeof(unsigned short);   // ~11.8 MB
    unsigned short* bPack = (unsigned short*)(ws + off);       // qk, bf16 B-layout
    off += (size_t)BB * QT * 2 * 32 * 16 * sizeof(unsigned short);   // ~295 KB
    float* bias = (float*)(ws + off);
    off += (size_t)BB * NE * sizeof(float);                          // ~369 KB
    float* candVal = (float*)(ws + off);
    off += (size_t)BB * NQ * NSLICES * TOPK * sizeof(float);         // ~2.95 MB
    int* candIdx = (int*)(ws + off);
    off += (size_t)BB * NQ * NSLICES * TOPK * sizeof(int);           // ~2.95 MB

    pack_mk_kernel<<<BB * NT, 64, 0, stream>>>(mk, aPack);
    pack_qk_kernel<<<BB * QT, 64, 0, stream>>>(qk, bPack);
    bias_kernel<<<(BB * NE + 255) / 256, 256, 0, stream>>>(mk, bias);
    aff_topk_kernel<<<BB * QT * NSLICES, 256, 0, stream>>>(aPack, bPack, bias,
                                                           candVal, candIdx);
    readout_kernel<<<BB * NQ, 256, 0, stream>>>(candVal, candIdx, mv, out);
}
